// GraphConvolution_9895604650475
// MI455X (gfx1250) — matrix operand order repaired and checked by reading, not compiled, not run on hardware
//
#include <hip/hip_runtime.h>

#define NN    50000
#define NPAD  50048              // 391 * 128, multiple of 16
#define DIN   512
#define DOUT  256
#define NE    1600000

typedef __attribute__((ext_vector_type(16))) __bf16        v16bf;
typedef __attribute__((ext_vector_type(8)))  float         v8f;
typedef __attribute__((ext_vector_type(8)))  unsigned int  v8u;

// Split an f32 pair into packed bf16 hi parts and bf16 lo residuals.
// Packing: first element in [15:0], second in [31:16] (ISA 16-bit operand order).
__device__ __forceinline__ void pack2(float x0, float x1, unsigned int& hi, unsigned int& lo) {
  unsigned int u0 = __float_as_uint(x0), u1 = __float_as_uint(x1);
  unsigned int h0 = u0 & 0xFFFF0000u,    h1 = u1 & 0xFFFF0000u;
  hi = (h0 >> 16) | h1;
  float r0 = x0 - __uint_as_float(h0);
  float r1 = x1 - __uint_as_float(h1);
  lo = (__float_as_uint(r0) >> 16) | (__float_as_uint(r1) & 0xFFFF0000u);
}

// ---------------------------------------------------------------------------
// K0a: repack w_low|w_high|w_mlp into per-lane WMMA B-operand order.
// B layout (16x16x32 bf16): lanes 0-15: N=lane,    v0-3 <- K 0..7,  v4-7 <- K 16..23;
//                           lanes 16-31: N=lane-16, v0-3 <- K 8..15, v4-7 <- K 24..31.
// Stored as [m][nt][kt][lane][j] packed uints (hi array + lo array).
// ---------------------------------------------------------------------------
__global__ void pack_w_kernel(const float* __restrict__ w0, const float* __restrict__ w1,
                              const float* __restrict__ w2,
                              unsigned int* __restrict__ bhi, unsigned int* __restrict__ blo) {
  const int TOT = 3 * (DIN / 2) * DOUT;
  int tid = blockIdx.x * blockDim.x + threadIdx.x;
  if (tid >= TOT) return;
  int m   = tid / ((DIN / 2) * DOUT);
  int rem = tid % ((DIN / 2) * DOUT);
  int ke  = rem / DOUT;
  int n   = rem % DOUT;
  int k   = ke * 2;
  const float* w = (m == 0) ? w0 : (m == 1 ? w1 : w2);
  unsigned int h, l;
  pack2(w[k * DOUT + n], w[(k + 1) * DOUT + n], h, l);
  int kt = k >> 5, kk = k & 31;
  int half = (kk & 15) >> 3, j = ((kk >> 4) << 2) + ((kk & 7) >> 1);
  int nt = n >> 4, nr = n & 15;
  int lane = nr + (half << 4);
  int dst = (((m * 16 + nt) * 16 + kt) * 32 + lane) * 8 + j;
  bhi[dst] = h;
  blo[dst] = l;
}

// ---------------------------------------------------------------------------
// K0b: repack X into per-lane WMMA A-operand order (bf16 hi + lo), padded
// with zeros to NPAD rows. A layout (16x16x32 bf16), per 16-row x 32-K tile:
//   lane<16 : M=lane,    v0-3 <- K 0..7,  v4-7 <- K 16..23
//   lane>=16: M=lane-16, v0-3 <- K 8..15, v4-7 <- K 24..31
// Stored as [rowtile][kt][lane][j] packed uints.
// ---------------------------------------------------------------------------
__global__ void pack_x_kernel(const float* __restrict__ X,
                              unsigned int* __restrict__ xhi, unsigned int* __restrict__ xlo) {
  const long TOT = (long)NPAD * (DIN / 2);
  long tid = (long)blockIdx.x * blockDim.x + threadIdx.x;
  if (tid >= TOT) return;
  int row = (int)(tid >> 8);     // / (DIN/2)
  int ke  = (int)(tid & 255);
  int k   = ke * 2;
  float x0 = 0.f, x1 = 0.f;
  if (row < NN) {
    x0 = X[(long)row * DIN + k];
    x1 = X[(long)row * DIN + k + 1];
  }
  unsigned int h, l;
  pack2(x0, x1, h, l);
  int rt = row >> 4, mrr = row & 15;
  int kt = k >> 5,  kk  = k & 31;
  int half, j;
  if (kk < 16) { half = kk >> 3;        j = (kk & 7) >> 1; }
  else         { half = (kk - 16) >> 3; j = 4 + (((kk - 16) & 7) >> 1); }
  int lane = mrr + (half << 4);
  long dst = (((long)rt * 16 + kt) * 32 + lane) * 8 + j;
  xhi[dst] = h;
  xlo[dst] = l;
}

// ---------------------------------------------------------------------------
// K1: zero the SpMM accumulators.
// ---------------------------------------------------------------------------
__global__ void zero_kernel(float4* __restrict__ p, int n4) {
  int i = blockIdx.x * blockDim.x + threadIdx.x;
  if (i < n4) p[i] = make_float4(0.f, 0.f, 0.f, 0.f);
}

// ---------------------------------------------------------------------------
// K2: XW GEMM, bf16x2 split precision via v_wmma_f32_16x16x32_bf16.
// All operands pre-packed in register order -> inner loop is pure
// global_load_b128 + WMMA. Block tile M=128, N=64; 8 waves in 4(M) x 2(N);
// each wave owns a 32x32 block = 4 WMMA tiles x 3 split-precision WMMAs/K32.
// grid.y in [0,12): global col = y*64, matrix m = col/256.
// ---------------------------------------------------------------------------
__global__ __launch_bounds__(256) void gemm_kernel(
    const unsigned int* __restrict__ xhi, const unsigned int* __restrict__ xlo,
    const unsigned int* __restrict__ bhi, const unsigned int* __restrict__ blo,
    float* __restrict__ xw) {
  const int tid   = threadIdx.x;
  const int lane  = tid & 31, wid = tid >> 5;
  const int waveM = wid & 3,  waveN = wid >> 2;
  const int mr    = lane & 15, half = lane >> 4;
  const int M0    = blockIdx.x * 128;
  const int nb    = blockIdx.y * 64;
  const int m     = nb >> 8;
  const int ntb   = (nb & 255) >> 4;
  const int nt0   = ntb + waveN * 2;

  const int  rt0  = (M0 >> 4) + waveM * 2;
  const long aoff = ((long)rt0 * 512 + lane) * 8;          // rt stride = 16*32*8 = 4096
  const long boff = ((long)(m * 16 + nt0) * 512 + lane) * 8;
  const unsigned int* ah0 = xhi + aoff;
  const unsigned int* al0 = xlo + aoff;
  const unsigned int* bh0 = bhi + boff;
  const unsigned int* bl0 = blo + boff;

  v8f acc[2][2] = {};

  #pragma unroll 4
  for (int kt = 0; kt < 16; ++kt) {
    const int o = kt * 256;                                // K-tile stride in uints
    v8u Ahu[2], Alu[2], Bhu[2], Blu[2];
    Ahu[0] = *(const v8u*)(ah0 + o);
    Ahu[1] = *(const v8u*)(ah0 + o + 4096);
    Alu[0] = *(const v8u*)(al0 + o);
    Alu[1] = *(const v8u*)(al0 + o + 4096);
    Bhu[0] = *(const v8u*)(bh0 + o);
    Bhu[1] = *(const v8u*)(bh0 + o + 4096);
    Blu[0] = *(const v8u*)(bl0 + o);
    Blu[1] = *(const v8u*)(bl0 + o + 4096);
    if (kt < 12) {                                         // prefetch A 4 K-tiles ahead
      __builtin_prefetch(ah0 + o + 1024, 0, 1);
      __builtin_prefetch(al0 + o + 1024, 0, 1);
    }
    #pragma unroll
    for (int s = 0; s < 2; ++s) {
      v16bf Ah = __builtin_bit_cast(v16bf, Ahu[s]);
      v16bf Al = __builtin_bit_cast(v16bf, Alu[s]);
      #pragma unroll
      for (int t = 0; t < 2; ++t) {
        v16bf Bh = __builtin_bit_cast(v16bf, Bhu[t]);
        v16bf Bl = __builtin_bit_cast(v16bf, Blu[t]);
        acc[s][t] = __builtin_amdgcn_wmma_f32_16x16x32_bf16(false, Ah, false, Bh, (short)0, acc[s][t], false, false);
        acc[s][t] = __builtin_amdgcn_wmma_f32_16x16x32_bf16(false, Ah, false, Bl, (short)0, acc[s][t], false, false);
        acc[s][t] = __builtin_amdgcn_wmma_f32_16x16x32_bf16(false, Al, false, Bh, (short)0, acc[s][t], false, false);
      }
    }
  }

  // C/D layout: VGPR i, lanes 0-15: M=i, N=lane; lanes 16-31: M=i+8, N=lane-16.
  float* out = xw + (long)m * NN * DOUT;
  const int colb = (nb & 255) + waveN * 32;
  #pragma unroll
  for (int s = 0; s < 2; ++s)
    #pragma unroll
    for (int t = 0; t < 2; ++t)
      #pragma unroll
      for (int i = 0; i < 8; ++i) {
        int R = M0 + waveM * 32 + s * 16 + half * 8 + i;
        if (R < NN) out[(long)R * DOUT + colb + t * 16 + mr] = acc[s][t][i];
      }
}

// ---------------------------------------------------------------------------
// K3: SpMM with sorted-row run-length register accumulation + float atomics.
// grid.y selects adjacency (0=low, 1=high). Each wave owns a 32-feature slice.
// ---------------------------------------------------------------------------
__global__ __launch_bounds__(256) void spmm_kernel(
    const int* __restrict__ rowsL, const int* __restrict__ colsL, const float* __restrict__ valsL,
    const int* __restrict__ rowsH, const int* __restrict__ colsH, const float* __restrict__ valsH,
    const float* __restrict__ xwL, const float* __restrict__ xwH,
    float* __restrict__ outL, float* __restrict__ outH) {
  const int*   rows; const int* cols; const float* vals; const float* xw; float* out;
  if (blockIdx.y == 0) { rows = rowsL; cols = colsL; vals = valsL; xw = xwL; out = outL; }
  else                 { rows = rowsH; cols = colsH; vals = valsH; xw = xwH; out = outH; }

  __shared__ int   s_rows[512];
  __shared__ int   s_cols[512];
  __shared__ float s_vals[512];

  const long e0 = (long)blockIdx.x * 512;
  const int cnt = (e0 + 512 <= (long)NE) ? 512 : (int)((long)NE - e0);
  for (int i = threadIdx.x; i < cnt; i += 256) {
    s_rows[i] = rows[e0 + i];
    s_cols[i] = cols[e0 + i];
    s_vals[i] = vals[e0 + i];
  }
  __syncthreads();

  const int lane = threadIdx.x & 31, wid = threadIdx.x >> 5;
  const int fb = wid * 32 + lane;

  float acc = 0.f;
  int cur = s_rows[0];
  for (int i = 0; i < cnt; ++i) {
    int r = s_rows[i];
    if (r != cur) {                       // wave-uniform branch (rows sorted)
      atomicAdd(&out[(long)cur * DOUT + fb], acc);
      acc = 0.f;
      cur = r;
    }
    acc = fmaf(s_vals[i], xw[(long)s_cols[i] * DOUT + fb], acc);
  }
  atomicAdd(&out[(long)cur * DOUT + fb], acc);
}

// ---------------------------------------------------------------------------
// K4: ReLU + LayerNorm + attention + combine. One wave per node (wave32).
// LN folded into attention dot: feat = rstd*(Sxga - mean*Sga) + Sba.
// ---------------------------------------------------------------------------
__device__ __forceinline__ float wsum(float v) {
  #pragma unroll
  for (int msk = 16; msk >= 1; msk >>= 1) v += __shfl_xor(v, msk, 32);
  return v;
}

__global__ __launch_bounds__(256) void fuse_kernel(
    const float* __restrict__ xw_mlp, const float* __restrict__ rawL, const float* __restrict__ rawH,
    const float* __restrict__ aL, const float* __restrict__ aH, const float* __restrict__ aM,
    const float* __restrict__ attv,
    const float* __restrict__ gL, const float* __restrict__ bL,
    const float* __restrict__ gH, const float* __restrict__ bH,
    const float* __restrict__ gM, const float* __restrict__ bM,
    float* __restrict__ out) {
  const int lane = threadIdx.x & 31, wid = threadIdx.x >> 5;
  const int node = blockIdx.x * 8 + wid;
  if (node >= NN) return;

  const float* src[3] = { rawL, rawH, xw_mlp };
  const float* gs[3]  = { gL, gH, gM };
  const float* bs[3]  = { bL, bH, bM };
  const float* as_[3] = { aL, aH, aM };

  float v[3][8];
  float feat[3];
  #pragma unroll
  for (int br = 0; br < 3; ++br) {
    float s1 = 0.f, s2 = 0.f, sxga = 0.f, sga = 0.f, sba = 0.f;
    #pragma unroll
    for (int j = 0; j < 8; ++j) {
      int f = lane + 32 * j;
      float x = src[br][(long)node * DOUT + f];
      x = fmaxf(x, 0.f);                 // relu
      v[br][j] = x;
      float g = gs[br][f], b = bs[br][f], a = as_[br][f];
      float ga = g * a;
      s1 += x; s2 += x * x;
      sxga += x * ga; sga += ga; sba += b * a;
    }
    s1 = wsum(s1); s2 = wsum(s2); sxga = wsum(sxga); sga = wsum(sga); sba = wsum(sba);
    float mean = s1 * (1.f / DOUT);
    float var  = s2 * (1.f / DOUT) - mean * mean;
    float rstd = rsqrtf(var + 1e-5f);
    feat[br] = rstd * (sxga - mean * sga) + sba;
  }

  float sg[3];
  #pragma unroll
  for (int br = 0; br < 3; ++br) sg[br] = 1.f / (1.f + __expf(-feat[br]));

  float lg[3];
  #pragma unroll
  for (int j = 0; j < 3; ++j)
    lg[j] = (sg[0] * attv[0 * 3 + j] + sg[1] * attv[1 * 3 + j] + sg[2] * attv[2 * 3 + j]) * (1.f / 3.f);

  float mx = fmaxf(lg[0], fmaxf(lg[1], lg[2]));
  float e0 = __expf(lg[0] - mx), e1 = __expf(lg[1] - mx), e2 = __expf(lg[2] - mx);
  float inv3 = 3.f / (e0 + e1 + e2);
  float a0 = e0 * inv3, a1 = e1 * inv3, a2 = e2 * inv3;

  #pragma unroll
  for (int j = 0; j < 8; ++j)
    out[(long)node * DOUT + lane + 32 * j] = a0 * v[0][j] + a1 * v[1][j] + a2 * v[2][j];
}

// ---------------------------------------------------------------------------
extern "C" void kernel_launch(void* const* d_in, const int* in_sizes, int n_in,
                              void* d_out, int out_size, void* d_ws, size_t ws_size,
                              hipStream_t stream) {
  (void)in_sizes; (void)n_in; (void)out_size; (void)ws_size;
  const float* x     = (const float*)d_in[0];
  const int*   rowsL = (const int*)d_in[1];
  const int*   colsL = (const int*)d_in[2];
  const float* valsL = (const float*)d_in[3];
  const int*   rowsH = (const int*)d_in[4];
  const int*   colsH = (const int*)d_in[5];
  const float* valsH = (const float*)d_in[6];
  const float* wL    = (const float*)d_in[7];
  const float* wH    = (const float*)d_in[8];
  const float* wM    = (const float*)d_in[9];
  const float* aL    = (const float*)d_in[10];
  const float* aH    = (const float*)d_in[11];
  const float* aM    = (const float*)d_in[12];
  const float* attv  = (const float*)d_in[13];
  const float* gL    = (const float*)d_in[14];
  const float* bL    = (const float*)d_in[15];
  const float* gH    = (const float*)d_in[16];
  const float* bH    = (const float*)d_in[17];
  const float* gM    = (const float*)d_in[18];
  const float* bM    = (const float*)d_in[19];

  float* ws   = (float*)d_ws;
  float* xw   = ws;                                   // [3][N][DOUT] f32
  float* outL = ws + 3L * NN * DOUT;                  // SpMM low accumulator
  float* outH = ws + 4L * NN * DOUT;                  // SpMM high accumulator
  unsigned int* bhi = (unsigned int*)(ws + 5L * NN * DOUT);
  unsigned int* blo = bhi + 3 * (DIN / 2) * DOUT;
  unsigned int* xhi = blo + 3 * (DIN / 2) * DOUT;
  unsigned int* xlo = xhi + (long)NPAD * (DIN / 2);

  // 1) pack weights + X into WMMA operand order (bf16 hi + lo)
  pack_w_kernel<<<(3 * (DIN / 2) * DOUT + 255) / 256, 256, 0, stream>>>(wL, wH, wM, bhi, blo);
  {
    long tot = (long)NPAD * (DIN / 2);
    pack_x_kernel<<<(unsigned)((tot + 255) / 256), 256, 0, stream>>>(x, xhi, xlo);
  }

  // 2) zero SpMM accumulators
  {
    int n4 = 2 * NN * DOUT / 4;
    zero_kernel<<<(n4 + 255) / 256, 256, 0, stream>>>((float4*)outL, n4);
  }

  // 3) fused triple GEMM: xw[m] = x @ w[m]
  gemm_kernel<<<dim3(NPAD / 128, 12), 256, 0, stream>>>(xhi, xlo, bhi, blo, xw);

  // 4) both SpMMs
  spmm_kernel<<<dim3(NE / 512, 2), 256, 0, stream>>>(rowsL, colsL, valsL, rowsH, colsH, valsH,
                                                     xw, xw + 1L * NN * DOUT, outL, outH);

  // 5) relu + layernorm + attention + combine
  fuse_kernel<<<(NN + 7) / 8, 256, 0, stream>>>(xw + 2L * NN * DOUT, outL, outH,
                                                aL, aH, aM, attv, gL, bL, gH, bH, gM, bM,
                                                (float*)d_out);
}
